// DecoderAutoregAdaIN_90658169684688
// MI455X (gfx1250) — matrix operand order repaired and checked by reading, not compiled, not run on hardware
//
#include <hip/hip_runtime.h>
#include <hip/hip_bf16.h>
#include <math.h>

// Problem constants
#define BB 2
#define TT 64
#define DD 512
#define MM 64
#define HH 8
#define LL 3
#define DFF 2048
#define HDIM 64
#define PERIOD 30
#define EPSV 1e-5f

typedef __attribute__((ext_vector_type(16))) __bf16 v16bf;
typedef __attribute__((ext_vector_type(8)))  float  v8f;

__device__ __forceinline__ float dot4(float4 a, float4 b) {
  return a.x * b.x + a.y * b.y + a.z * b.z + a.w * b.w;
}

// ---------------------------------------------------------------------------
// WMMA GEMM: C[Mrows,N] = A[Mrows,K] @ W[N,K]^T + bias   (bf16 in, f32 acc)
// Mrows%16==0, N%16==0, K%32==0. One 16x16 output tile per wave.
// Fragment fills use b128 loads: per k-tile each lane's A data is two
// contiguous 8-float runs, B is likewise two contiguous 8-float runs.
// ---------------------------------------------------------------------------
__global__ void gemm_wmma_k(const float* __restrict__ A, const float* __restrict__ W,
                            const float* __restrict__ bias, float* __restrict__ C,
                            int Mrows, int N, int K) {
  int wave = (int)((blockIdx.x * blockDim.x + threadIdx.x) >> 5);
  int lane = (int)(threadIdx.x & 31);
  int tiles_n = N >> 4;
  int tiles = (Mrows >> 4) * tiles_n;
  if (wave >= tiles) return;                 // wave-uniform: EXEC stays all-ones
  int tm = wave / tiles_n, tn = wave % tiles_n;
  int half = lane >> 4;
  int l15  = lane & 15;
  const float* arow = A + (size_t)(tm * 16 + l15) * K;
  const float* wrow = W + (size_t)(tn * 16 + l15) * K;
  v8f acc = {};
  for (int k0 = 0; k0 < K; k0 += 32) {
    // A 16x32 bf16 layout: lanes 0-15 K={0..7,16..23}, lanes 16-31 K={8..15,24..31}
    const float4* a0 = (const float4*)(arow + k0 + 8 * half);
    const float4* a1 = (const float4*)(arow + k0 + 16 + 8 * half);
    // B 32x16 bf16 layout: lanes 0-15 K={0..7,16..23}, lanes 16-31 K={8..15,24..31}
    const float4* b0 = (const float4*)(wrow + k0 + 8 * half);
    float4 A0 = a0[0], A1 = a0[1], A2 = a1[0], A3 = a1[1];
    float4 B0 = b0[0], B1 = b0[1], B2 = b0[4], B3 = b0[5];
    v16bf af, bfv;
    af[0]  = (__bf16)A0.x; af[1]  = (__bf16)A0.y; af[2]  = (__bf16)A0.z; af[3]  = (__bf16)A0.w;
    af[4]  = (__bf16)A1.x; af[5]  = (__bf16)A1.y; af[6]  = (__bf16)A1.z; af[7]  = (__bf16)A1.w;
    af[8]  = (__bf16)A2.x; af[9]  = (__bf16)A2.y; af[10] = (__bf16)A2.z; af[11] = (__bf16)A2.w;
    af[12] = (__bf16)A3.x; af[13] = (__bf16)A3.y; af[14] = (__bf16)A3.z; af[15] = (__bf16)A3.w;
    bfv[0]  = (__bf16)B0.x; bfv[1]  = (__bf16)B0.y; bfv[2]  = (__bf16)B0.z; bfv[3]  = (__bf16)B0.w;
    bfv[4]  = (__bf16)B1.x; bfv[5]  = (__bf16)B1.y; bfv[6]  = (__bf16)B1.z; bfv[7]  = (__bf16)B1.w;
    bfv[8]  = (__bf16)B2.x; bfv[9]  = (__bf16)B2.y; bfv[10] = (__bf16)B2.z; bfv[11] = (__bf16)B2.w;
    bfv[12] = (__bf16)B3.x; bfv[13] = (__bf16)B3.y; bfv[14] = (__bf16)B3.z; bfv[15] = (__bf16)B3.w;
    acc = __builtin_amdgcn_wmma_f32_16x16x32_bf16(false, af, false, bfv,
                                                  (short)0, acc, false, false);
  }
#pragma unroll
  for (int r = 0; r < 8; ++r) {
    int m = tm * 16 + half * 8 + r;   // C: VGPR r -> M=r (lanes 0-15) / M=8+r (16-31)
    int n = tn * 16 + l15;
    C[(size_t)m * N + n] = acc[r] + bias[n];
  }
}

// ---------------------------------------------------------------------------
// Generic batched GEMV: y[b*ybs + o] = dot(x[b*xbs..], W[o,:]) + bias, wave/out.
// float4 loads: a wave pulls 512B contiguous per issue (coalesced).
// ---------------------------------------------------------------------------
__global__ void gemv_k(const float* __restrict__ x, int xbs,
                       const float* __restrict__ W, const float* __restrict__ bias,
                       float* __restrict__ y, int ybs,
                       int In, int Out, int relu, int Bn) {
  int wid  = (int)((blockIdx.x * blockDim.x + threadIdx.x) >> 5);
  int lane = (int)(threadIdx.x & 31);
  int b = wid / Out, o = wid % Out;
  if (b >= Bn) return;
  const float4* xr = (const float4*)(x + (size_t)b * xbs);
  const float4* wr = (const float4*)(W + (size_t)o * In);
  int nv = In >> 2;
  float s = 0.f;
  for (int i = lane; i < nv; i += 32) s += dot4(xr[i], wr[i]);
#pragma unroll
  for (int off = 16; off > 0; off >>= 1) s += __shfl_xor(s, off, 32);
  if (lane == 0) {
    float v = s + bias[o];
    if (relu) v = fmaxf(v, 0.f);
    y[(size_t)b * ybs + o] = v;
  }
}

// QKV projection for one token; K,V appended to caches at position t.
__global__ void qkv_k(const float* __restrict__ x, const float* __restrict__ W,
                      const float* __restrict__ bias, float* __restrict__ q,
                      float* __restrict__ kc, float* __restrict__ vc, int t) {
  int wid  = (int)((blockIdx.x * blockDim.x + threadIdx.x) >> 5);
  int lane = (int)(threadIdx.x & 31);
  int b = wid / (3 * DD), o = wid % (3 * DD);
  if (b >= BB) return;
  const float4* xr = (const float4*)(x + (size_t)b * DD);
  const float4* wr = (const float4*)(W + (size_t)o * DD);
  float s = 0.f;
#pragma unroll 4
  for (int i = lane; i < (DD >> 2); i += 32) s += dot4(xr[i], wr[i]);
#pragma unroll
  for (int off = 16; off > 0; off >>= 1) s += __shfl_xor(s, off, 32);
  if (lane == 0) {
    float v = s + bias[o];
    int seg = o >> 9, d = o & (DD - 1);
    if (seg == 0)      q[b * DD + d] = v;
    else if (seg == 1) kc[((size_t)b * TT + t) * DD + d] = v;
    else               vc[((size_t)b * TT + t) * DD + d] = v;
  }
}

// Single-token causal attention with periodic-ALiBi bias. grid = B*H, block = 64.
__global__ void attn_k(const float* __restrict__ q, const float* __restrict__ kc,
                       const float* __restrict__ vc, float* __restrict__ out, int t) {
  int b = (int)(blockIdx.x >> 3), h = (int)(blockIdx.x & 7);
  int j = (int)threadIdx.x;
  __shared__ float qs[HDIM], sc[TT], red[TT];
  qs[j] = q[b * DD + h * HDIM + j] * 0.125f;   // 1/sqrt(64)
  __syncthreads();
  float s = -1e30f;
  if (j <= t) {
    const float4* kr = (const float4*)(kc + ((size_t)b * TT + j) * DD + h * HDIM);
    const float4* q4 = (const float4*)qs;
    float acc = 0.f;
#pragma unroll
    for (int d = 0; d < (HDIM >> 2); ++d) acc += dot4(q4[d], kr[d]);
    float slope = exp2f(-(float)(h + 1));
    acc -= slope * (float)((t - j) / PERIOD);
    s = acc;
  }
  sc[j] = s; red[j] = s; __syncthreads();
  for (int off = 32; off > 0; off >>= 1) {
    if (j < off) red[j] = fmaxf(red[j], red[j + off]);
    __syncthreads();
  }
  float mx = red[0]; __syncthreads();
  float p = (j <= t) ? __expf(sc[j] - mx) : 0.f;
  sc[j] = p; red[j] = p; __syncthreads();
  for (int off = 32; off > 0; off >>= 1) {
    if (j < off) red[j] += red[j + off];
    __syncthreads();
  }
  float inv = 1.f / red[0];
  float o = 0.f;
  const float* vbase = vc + (size_t)b * TT * DD + h * HDIM + j;
  for (int jj = 0; jj <= t; ++jj) o += sc[jj] * vbase[(size_t)jj * DD];
  out[b * DD + h * HDIM + j] = o * inv;
}

// out = LayerNorm(a + r) * g + beta   (grid = B, block = 512)
__global__ void addln_k(const float* __restrict__ a, const float* __restrict__ r, int rbs,
                        const float* __restrict__ g, const float* __restrict__ be,
                        float* __restrict__ out) {
  int b = (int)blockIdx.x, d = (int)threadIdx.x;
  float v = a[b * DD + d] + r[(size_t)b * rbs + d];
  __shared__ float red[DD];
  red[d] = v; __syncthreads();
  for (int off = 256; off > 0; off >>= 1) {
    if (d < off) red[d] += red[d + off];
    __syncthreads();
  }
  float mu = red[0] * (1.f / DD); __syncthreads();
  float c = v - mu;
  red[d] = c * c; __syncthreads();
  for (int off = 256; off > 0; off >>= 1) {
    if (d < off) red[d] += red[d + off];
    __syncthreads();
  }
  float var = red[0] * (1.f / DD);
  out[b * DD + d] = c * rsqrtf(var + EPSV) * g[d] + be[d];
}

// x2 = LN2( LN1(a + r) + cross )  — fuses self-attn LN and cross-add LN.
__global__ void addln2_k(const float* __restrict__ a, const float* __restrict__ r,
                         const float* __restrict__ crossp, int cbs,
                         const float* __restrict__ g1, const float* __restrict__ b1,
                         const float* __restrict__ g2, const float* __restrict__ b2,
                         float* __restrict__ out) {
  int b = (int)blockIdx.x, d = (int)threadIdx.x;
  __shared__ float red[DD];
  // LN1
  float v = a[b * DD + d] + r[b * DD + d];
  red[d] = v; __syncthreads();
  for (int off = 256; off > 0; off >>= 1) {
    if (d < off) red[d] += red[d + off];
    __syncthreads();
  }
  float mu = red[0] * (1.f / DD); __syncthreads();
  float c = v - mu;
  red[d] = c * c; __syncthreads();
  for (int off = 256; off > 0; off >>= 1) {
    if (d < off) red[d] += red[d + off];
    __syncthreads();
  }
  float x1 = c * rsqrtf(red[0] * (1.f / DD) + EPSV) * g1[d] + b1[d];
  __syncthreads();
  // LN2 over (x1 + cross)
  float v2 = x1 + crossp[(size_t)b * cbs + d];
  red[d] = v2; __syncthreads();
  for (int off = 256; off > 0; off >>= 1) {
    if (d < off) red[d] += red[d + off];
    __syncthreads();
  }
  float mu2 = red[0] * (1.f / DD); __syncthreads();
  float c2 = v2 - mu2;
  red[d] = c2 * c2; __syncthreads();
  for (int off = 256; off > 0; off >>= 1) {
    if (d < off) red[d] += red[d + off];
    __syncthreads();
  }
  out[b * DD + d] = c2 * rsqrtf(red[0] * (1.f / DD) + EPSV) * g2[d] + b2[d];
}

// Instance-norm stats over time axis: mu/var per (b,d)
__global__ void instnorm_stats_k(const float* __restrict__ c, float* __restrict__ mu,
                                 float* __restrict__ var) {
  int idx = (int)(blockIdx.x * blockDim.x + threadIdx.x);
  if (idx >= BB * DD) return;
  int b = idx >> 9, d = idx & (DD - 1);
  float s = 0.f, s2 = 0.f;
  for (int t = 0; t < TT; ++t) {
    float v = c[((size_t)b * TT + t) * DD + d];
    s += v; s2 += v * v;
  }
  float m = s * (1.f / TT);
  mu[idx] = m; var[idx] = s2 * (1.f / TT) - m * m;
}

// memory = gamma * normed + beta   (style = [B, 2D])
__global__ void adain_mod_k(const float* __restrict__ c, const float* __restrict__ mu,
                            const float* __restrict__ var, const float* __restrict__ style,
                            float* __restrict__ mem) {
  int idx = (int)(blockIdx.x * blockDim.x + threadIdx.x);
  if (idx >= BB * TT * DD) return;
  int b = idx / (TT * DD);
  int d = idx & (DD - 1);
  float nv = (c[idx] - mu[b * DD + d]) * rsqrtf(var[b * DD + d] + EPSV);
  mem[idx] = style[b * 2 * DD + d] * nv + style[b * 2 * DD + DD + d];
}

// x_tok = src @ mm_w^T + mm_b + PE(pos)   (src is [B, M] with b-stride sbs)
__global__ void mm_pe_k(const float* __restrict__ src, int sbs,
                        const float* __restrict__ mm_w, const float* __restrict__ mm_b,
                        float* __restrict__ x_tok, int pos) {
  int wid  = (int)((blockIdx.x * blockDim.x + threadIdx.x) >> 5);
  int lane = (int)(threadIdx.x & 31);
  int b = wid >> 9, d = wid & (DD - 1);
  if (b >= BB) return;
  const float4* s4 = (const float4*)(src + (size_t)b * sbs);
  const float4* w4 = (const float4*)(mm_w + (size_t)d * MM);
  float acc = 0.f;
  if (lane < (MM >> 2)) acc = dot4(s4[lane], w4[lane]);
#pragma unroll
  for (int off = 16; off > 0; off >>= 1) acc += __shfl_xor(acc, off, 32);
  if (lane == 0) {
    int p = pos % PERIOD;
    float div = __expf(-(float)(d & ~1) * (9.210340371976184f / (float)DD));
    float ang = (float)p * div;
    float pe = (d & 1) ? __cosf(ang) : __sinf(ang);
    x_tok[b * DD + d] = acc + mm_b[d] + pe;
  }
}

extern "C" void kernel_launch(void* const* d_in, const int* in_sizes, int n_in,
                              void* d_out, int out_size, void* d_ws, size_t ws_size,
                              hipStream_t stream) {
  const float* content   = (const float*)d_in[0];
  const float* style_cod = (const float*)d_in[1];
  const float* init_st   = (const float*)d_in[2];
  const float* adain_w   = (const float*)d_in[3];
  const float* adain_b   = (const float*)d_in[4];
  const float* mm_w      = (const float*)d_in[5];
  const float* mm_b      = (const float*)d_in[6];
  const float* mmr_w     = (const float*)d_in[7];
  const float* mmr_b     = (const float*)d_in[8];
  const float* sa_w      = (const float*)d_in[9];
  const float* sa_b      = (const float*)d_in[10];
  const float* sa_o_w    = (const float*)d_in[11];
  const float* sa_o_b    = (const float*)d_in[12];
  const float* ca_w      = (const float*)d_in[13];
  const float* ca_b      = (const float*)d_in[14];
  const float* ca_o_w    = (const float*)d_in[15];
  const float* ca_o_b    = (const float*)d_in[16];
  const float* ff1_w     = (const float*)d_in[17];
  const float* ff1_b     = (const float*)d_in[18];
  const float* ff2_w     = (const float*)d_in[19];
  const float* ff2_b     = (const float*)d_in[20];
  const float* ln_g      = (const float*)d_in[21];
  const float* ln_b      = (const float*)d_in[22];
  float* out = (float*)d_out;

  float* ws = (float*)d_ws;
  size_t off = 0;
  auto alloc = [&](size_t n) { float* p = ws + off; off += n; return p; };
  const size_t BTD = (size_t)BB * TT * DD;     // 65536
  float* style = alloc(BB * 2 * DD);
  float* mu    = alloc(BB * DD);
  float* var   = alloc(BB * DD);
  float* mem   = alloc(BTD);
  float* cv    = alloc(BTD);
  float* cross = alloc(LL * BTD);
  float* kc    = alloc(LL * BTD);
  float* vc    = alloc(LL * BTD);
  float* x_tok = alloc(BB * DD);
  float* qb    = alloc(BB * DD);
  float* attn  = alloc(BB * DD);
  float* tmp   = alloc(BB * DD);
  float* x2    = alloc(BB * DD);
  float* hb    = alloc(BB * DFF);
  float* x3    = alloc(BB * DD);

  // ---- Phase 0: AdaIN memory + degenerate cross-attention precompute ----
  instnorm_stats_k<<<(BB * DD + 255) / 256, 256, 0, stream>>>(content, mu, var);
  gemv_k<<<(BB * 2 * DD) / 8, 256, 0, stream>>>(style_cod, DD, adain_w, adain_b,
                                                style, 2 * DD, DD, 2 * DD, 0, BB);
  adain_mod_k<<<(int)(BTD / 256), 256, 0, stream>>>(content, mu, var, style, mem);
  for (int l = 0; l < LL; ++l) {
    // v = memory @ Wv^T + bv ; cross_out = v @ Wo^T + bo (cross mask == identity)
    gemm_wmma_k<<<32, 256, 0, stream>>>(mem, ca_w + ((size_t)l * 3 * DD + 2 * DD) * DD,
                                        ca_b + (size_t)l * 3 * DD + 2 * DD,
                                        cv, BB * TT, DD, DD);
    gemm_wmma_k<<<32, 256, 0, stream>>>(cv, ca_o_w + (size_t)l * DD * DD,
                                        ca_o_b + (size_t)l * DD,
                                        cross + (size_t)l * BTD, BB * TT, DD, DD);
  }

  // ---- Token 0 embedding: init_state @ mm_w^T + mm_b + PE(0) ----
  mm_pe_k<<<(BB * DD) / 8, 256, 0, stream>>>(init_st, MM, mm_w, mm_b, x_tok, 0);

  // ---- 64 incremental decode steps (KV-cached; exact vs reference) ----
  for (int t = 0; t < TT; ++t) {
    for (int l = 0; l < LL; ++l) {
      const float* xi = (l == 0) ? x_tok : x3;
      qkv_k<<<(BB * 3 * DD) / 8, 256, 0, stream>>>(xi, sa_w + (size_t)l * 3 * DD * DD,
                                                   sa_b + (size_t)l * 3 * DD, qb,
                                                   kc + (size_t)l * BTD,
                                                   vc + (size_t)l * BTD, t);
      attn_k<<<BB * HH, TT, 0, stream>>>(qb, kc + (size_t)l * BTD,
                                         vc + (size_t)l * BTD, attn, t);
      gemv_k<<<(BB * DD) / 8, 256, 0, stream>>>(attn, DD, sa_o_w + (size_t)l * DD * DD,
                                                sa_o_b + (size_t)l * DD, tmp, DD,
                                                DD, DD, 0, BB);
      addln2_k<<<BB, DD, 0, stream>>>(xi, tmp,
                                      cross + (size_t)l * BTD + (size_t)t * DD, TT * DD,
                                      ln_g + (size_t)(l * 3 + 0) * DD,
                                      ln_b + (size_t)(l * 3 + 0) * DD,
                                      ln_g + (size_t)(l * 3 + 1) * DD,
                                      ln_b + (size_t)(l * 3 + 1) * DD, x2);
      gemv_k<<<(BB * DFF) / 8, 256, 0, stream>>>(x2, DD, ff1_w + (size_t)l * DFF * DD,
                                                 ff1_b + (size_t)l * DFF, hb, DFF,
                                                 DD, DFF, 1, BB);
      gemv_k<<<(BB * DD) / 8, 256, 0, stream>>>(hb, DFF, ff2_w + (size_t)l * DD * DFF,
                                                ff2_b + (size_t)l * DD, tmp, DD,
                                                DFF, DD, 0, BB);
      addln_k<<<BB, DD, 0, stream>>>(x2, tmp, DD, ln_g + (size_t)(l * 3 + 2) * DD,
                                     ln_b + (size_t)(l * 3 + 2) * DD, x3);
    }
    // output row t: x3 @ mmr_w^T + mmr_b -> d_out[:, t, :]
    gemv_k<<<(BB * MM) / 8, 256, 0, stream>>>(x3, DD, mmr_w, mmr_b,
                                              out + (size_t)t * MM, TT * MM,
                                              DD, MM, 0, BB);
    if (t < TT - 1) {
      // next token embedding: out_row @ mm_w^T + mm_b + PE(t+1)
      mm_pe_k<<<(BB * DD) / 8, 256, 0, stream>>>(out + (size_t)t * MM, TT * MM,
                                                 mm_w, mm_b, x_tok, t + 1);
    }
  }
}
